// TutteLayer_old_9371618640210
// MI455X (gfx1250) — compile-verified, matrix operand order checked by reading, not run on hardware
//
#include <hip/hip_runtime.h>
#include <hip/hip_bf16.h>
#include <math.h>

typedef __attribute__((ext_vector_type(16))) __bf16 v16bf;
typedef __attribute__((ext_vector_type(8)))  float  v8f;
typedef __attribute__((ext_vector_type(4)))  int    v4i;

#if defined(__HIP_DEVICE_COMPILE__) && __has_builtin(__builtin_amdgcn_global_load_async_to_lds_b128)
#define HAVE_ASYNC_LDS 1
typedef __attribute__((address_space(1))) v4i* gv4i_p;
typedef __attribute__((address_space(3))) v4i* lv4i_p;
#endif

// ---------------------------------------------------------------- utilities
__global__ void k_zero(float* p, int n) {
  for (int i = blockIdx.x * blockDim.x + threadIdx.x; i < n; i += gridDim.x * blockDim.x)
    p[i] = 0.0f;
}

__device__ __forceinline__ float sigmoidf_(float x) { return 1.0f / (1.0f + expf(-x)); }

// w = sigmoid(W)*0.6 + 0.2
__global__ void k_weights(const float* Wv, float* w, int E) {
  int e = blockIdx.x * blockDim.x + threadIdx.x;
  if (e < E) w[e] = sigmoidf_(Wv[e]) * 0.6f + 0.2f;
}

// boundary positions on unit square (single block; nB = 200 <= 256)
__global__ void k_boundary(const float* angle_var, const int* bound, int nB, float* posf) {
  __shared__ float sa[256];
  int t = threadIdx.x;
  float a = 0.0f;
  if (t < nB) a = sigmoidf_(angle_var[t]) * 0.6f + 0.2f;
  sa[t] = a;
  __syncthreads();
  for (int off = 1; off < 256; off <<= 1) {   // inclusive scan
    float x = (t >= off) ? sa[t - off] : 0.0f;
    __syncthreads();
    sa[t] += x;
    __syncthreads();
  }
  float total = sa[255];
  if (t < nB) {
    const float pi = 3.14159265358979323846f;
    float ang = sa[t] / total * (2.0f * pi);
    float tt = tanf(ang);
    bool m1 = (ang > 7.0f * pi / 4.0f) || (ang <= pi / 4.0f);
    bool m2 = (ang > pi / 4.0f) && (ang <= 3.0f * pi / 4.0f);
    bool m3 = (ang > 3.0f * pi / 4.0f) && (ang <= 5.0f * pi / 4.0f);
    float bx, by;
    if (m1)      { bx = 1.0f;       by = tt;    }
    else if (m2) { bx = 1.0f / tt;  by = 1.0f;  }
    else if (m3) { bx = -1.0f;      by = -tt;   }
    else         { bx = -1.0f / tt; by = -1.0f; }
    int v = bound[t];
    posf[v * 2 + 0] = bx;
    posf[v * 2 + 1] = by;
  }
}

// deg = segment_sum(w,row); b += w*is_bound(col)*pos[col]
__global__ void k_deg_b(const int* edges, const float* w, const int* imap,
                        const float* posf, float* deg, float* bvec, int E) {
  int e = blockIdx.x * blockDim.x + threadIdx.x;
  if (e >= E) return;
  int r = edges[e], c = edges[E + e];
  float we = w[e];
  atomicAdd(&deg[r], we);
  if (imap[c] < 0) {
    atomicAdd(&bvec[r * 2 + 0], we * posf[c * 2 + 0]);
    atomicAdd(&bvec[r * 2 + 1], we * posf[c * 2 + 1]);
  }
}

// A = 0; identity on padded diagonal rows
__global__ void k_initA(float* A, int NP, int nI) {
  size_t tot = (size_t)NP * NP;
  for (size_t i = blockIdx.x * (size_t)blockDim.x + threadIdx.x; i < tot;
       i += (size_t)gridDim.x * blockDim.x) {
    int r = (int)(i / NP), c = (int)(i % NP);
    A[i] = (r == c && r >= nI) ? 1.0f : 0.0f;
  }
}

__global__ void k_scatterA(const int* edges, const float* w, const int* imap,
                           float* A, int NP, int E) {
  int e = blockIdx.x * blockDim.x + threadIdx.x;
  if (e >= E) return;
  int ri = imap[edges[e]], ci = imap[edges[E + e]];
  if (ri >= 0 && ci >= 0) A[(size_t)ri * NP + ci] = -w[e];   // directed edges unique
}

__global__ void k_diagA(const int* iverts, const float* deg, float* A, int NP, int nI) {
  int i = blockIdx.x * blockDim.x + threadIdx.x;
  if (i < nI) A[(size_t)i * NP + i] = deg[iverts[i]];
}

// ---------------------------------------------------------------- blocked LU, NB = 32
__global__ void k_panel(float* A, int NP, int k0) {
  __shared__ float s[32][33];
  int t = threadIdx.x;
  for (int idx = t; idx < 1024; idx += 256) {
    int r = idx >> 5, c = idx & 31;
    s[r][c] = A[(size_t)(k0 + r) * NP + k0 + c];
  }
  __syncthreads();
  for (int p = 0; p < 31; ++p) {
    float piv = s[p][p];
    for (int i = p + 1 + t; i < 32; i += 256) s[i][p] /= piv;
    __syncthreads();
    for (int idx = t; idx < 1024; idx += 256) {
      int r = idx >> 5, c = idx & 31;
      if (r > p && c > p) s[r][c] -= s[r][p] * s[p][c];
    }
    __syncthreads();
  }
  for (int idx = t; idx < 1024; idx += 256) {
    int r = idx >> 5, c = idx & 31;
    A[(size_t)(k0 + r) * NP + k0 + c] = s[r][c];
  }
}

// L21 = A21 * U11^{-1}: one row per thread
__global__ void k_trsmL(float* A, int NP, int k0, int rem) {
  __shared__ float U[32][33];
  int t = threadIdx.x;
  for (int idx = t; idx < 1024; idx += 256) {
    int r = idx >> 5, c = idx & 31;
    U[r][c] = A[(size_t)(k0 + r) * NP + k0 + c];
  }
  __syncthreads();
  int row = blockIdx.x * blockDim.x + t;
  if (row >= rem) return;
  size_t base = (size_t)(k0 + 32 + row) * NP + k0;
  float r[32];
#pragma unroll
  for (int j = 0; j < 32; ++j) r[j] = A[base + j];
  for (int j = 0; j < 32; ++j) {
    float acc = r[j];
    for (int q = 0; q < j; ++q) acc -= r[q] * U[q][j];
    r[j] = acc / U[j][j];
  }
#pragma unroll
  for (int j = 0; j < 32; ++j) A[base + j] = r[j];
}

// U12 = L11^{-1} * A12: one column per thread (unit lower)
__global__ void k_trsmU(float* A, int NP, int k0, int rem) {
  __shared__ float L[32][33];
  int t = threadIdx.x;
  for (int idx = t; idx < 1024; idx += 256) {
    int r = idx >> 5, c = idx & 31;
    L[r][c] = A[(size_t)(k0 + r) * NP + k0 + c];
  }
  __syncthreads();
  int col = blockIdx.x * blockDim.x + t;
  if (col >= rem) return;
  size_t cg = (size_t)(k0 + 32 + col);
  float u[32];
#pragma unroll
  for (int i = 0; i < 32; ++i) u[i] = A[(size_t)(k0 + i) * NP + cg];
  for (int i = 1; i < 32; ++i) {
    float acc = u[i];
    for (int q = 0; q < i; ++q) acc -= L[i][q] * u[q];
    u[i] = acc;
  }
#pragma unroll
  for (int i = 0; i < 32; ++i) A[(size_t)(k0 + i) * NP + cg] = u[i];
}

// trailing update A22 -= L21*U12 via bf16 WMMA (fp32 accumulate).
// 256 threads = 8 waves compute a 64x64 tile; L21/U12 panels staged to LDS
// (async global->LDS b128 when available). Each wave: 1 A-frag, 2 WMMAs.
__global__ void __launch_bounds__(256) k_update(float* A, int NP, int k0, int rem) {
  __shared__ __align__(16) float Ls[64][36];   // 64 rows x 32 k   (+pad, 16B-aligned rows)
  __shared__ __align__(16) float Us[32][68];   // 32 k   x 64 cols (+pad, 16B-aligned rows)
  const int t = threadIdx.x;
  const int tr0 = k0 + 32;
  const int rbase = blockIdx.y * 64;
  const int cbase = blockIdx.x * 64;

  // ---- stage panels: 2048 floats each = 512 x b128; 256 threads x 2 quads per panel
#pragma unroll
  for (int i = 0; i < 2; ++i) {
    int idx = t + 256 * i;
    // L21 quad
    {
      int row = idx >> 3, q = idx & 7;              // 8 quads per 32-wide row
      bool ok = (rbase + row) < rem;                // cutoff is a multiple of 32
      float* lp = &Ls[row][q * 4];
      if (ok) {
        const float* gp = A + (size_t)(tr0 + rbase + row) * NP + k0 + q * 4;
#ifdef HAVE_ASYNC_LDS
        __builtin_amdgcn_global_load_async_to_lds_b128((gv4i_p)gp, (lv4i_p)lp, 0, 0);
#else
        lp[0] = gp[0]; lp[1] = gp[1]; lp[2] = gp[2]; lp[3] = gp[3];
#endif
      } else {
        lp[0] = 0.0f; lp[1] = 0.0f; lp[2] = 0.0f; lp[3] = 0.0f;
      }
    }
    // U12 quad
    {
      int row = idx >> 4, q = idx & 15;             // 16 quads per 64-wide row
      bool ok = (cbase + q * 4) < rem;              // cutoff is a multiple of 32
      float* lp = &Us[row][q * 4];
      if (ok) {
        const float* gp = A + (size_t)(k0 + row) * NP + tr0 + cbase + q * 4;
#ifdef HAVE_ASYNC_LDS
        __builtin_amdgcn_global_load_async_to_lds_b128((gv4i_p)gp, (lv4i_p)lp, 0, 0);
#else
        lp[0] = gp[0]; lp[1] = gp[1]; lp[2] = gp[2]; lp[3] = gp[3];
#endif
      } else {
        lp[0] = 0.0f; lp[1] = 0.0f; lp[2] = 0.0f; lp[3] = 0.0f;
      }
    }
  }
#ifdef HAVE_ASYNC_LDS
#if __has_builtin(__builtin_amdgcn_s_wait_asynccnt)
  __builtin_amdgcn_s_wait_asynccnt(0);
#else
  asm volatile("s_wait_asynccnt 0" ::: "memory");
#endif
#endif
  __syncthreads();

  // ---- compute: wave w handles subtiles (sr, sc) and (sr, sc+1)
  const int wave = t >> 5, lane = t & 31;
  const int lo = lane & 15, hi = lane >> 4;
  const int sr = wave >> 1;                 // 0..3
  const int sc = (wave & 1) * 2;            // 0 or 2
  if (rbase + sr * 16 >= rem) return;       // wave-uniform guard (cutoff multiple of 32)

  v16bf af;
#pragma unroll
  for (int e = 0; e < 16; ++e) {            // ISA 16-bit A 16x32 layout, A-neg folded in
    int v = e >> 1, odd = e & 1;
    int kk = (v < 4 ? (hi * 8 + 2 * v) : (16 + hi * 8 + 2 * (v - 4))) + odd;
    af[e] = (__bf16)(-Ls[sr * 16 + lo][kk]);
  }

#pragma unroll
  for (int s = 0; s < 2; ++s) {
    int sct = sc + s;
    if (cbase + sct * 16 >= rem) continue;  // wave-uniform guard
    const int r0 = tr0 + rbase + sr * 16;
    const int c0 = tr0 + cbase + sct * 16;
    v8f c;
#pragma unroll
    for (int v = 0; v < 8; ++v)             // ISA 16x16 f32 C/D layout
      c[v] = A[(size_t)(r0 + v + 8 * hi) * NP + c0 + lo];
    v16bf bfm;
#pragma unroll
    for (int e = 0; e < 16; ++e) {
      int v = e >> 1, odd = e & 1;
      int kk = (v < 4 ? (hi * 8 + 2 * v) : (16 + hi * 8 + 2 * (v - 4))) + odd;
      bfm[e] = (__bf16)Us[kk][sct * 16 + lo];
    }
    c = __builtin_amdgcn_wmma_f32_16x16x32_bf16(false, af, false, bfm,
                                                (short)0, c, false, false);
#pragma unroll
    for (int v = 0; v < 8; ++v)
      A[(size_t)(r0 + v + 8 * hi) * NP + c0 + lo] = c[v];
  }
}

// ---------------------------------------------------------------- solve helpers
__global__ void k_gather_rhs(const float* bvec, const int* iverts, float* x, int nI, int NP) {
  int i = blockIdx.x * blockDim.x + threadIdx.x;
  if (i >= NP) return;
  if (i < nI) {
    int v = iverts[i];
    x[i * 2 + 0] = bvec[v * 2 + 0];
    x[i * 2 + 1] = bvec[v * 2 + 1];
  } else {
    x[i * 2 + 0] = 0.0f;
    x[i * 2 + 1] = 0.0f;
  }
}

// in-place LU triangular solves, 2 RHS, single workgroup; RHS held in LDS
__global__ void k_solve(const float* A, int NP, float* v) {
  extern __shared__ float smem[];
  float* sv  = smem;            // 2*NP
  float* red = smem + 2 * NP;   // 512
  int t = threadIdx.x;
  for (int j = t; j < 2 * NP; j += 256) sv[j] = v[j];
  __syncthreads();
  // forward (unit lower)
  for (int i = 1; i < NP; ++i) {
    const float* row = A + (size_t)i * NP;
    float a0 = 0.0f, a1 = 0.0f;
    for (int j = t; j < i; j += 256) {
      float lij = row[j];
      a0 += lij * sv[j * 2]; a1 += lij * sv[j * 2 + 1];
    }
    red[2 * t] = a0; red[2 * t + 1] = a1;
    __syncthreads();
    for (int s = 128; s > 0; s >>= 1) {
      if (t < s) { red[2 * t] += red[2 * (t + s)]; red[2 * t + 1] += red[2 * (t + s) + 1]; }
      __syncthreads();
    }
    if (t == 0) { sv[i * 2] -= red[0]; sv[i * 2 + 1] -= red[1]; }
    __syncthreads();
  }
  // backward
  for (int i = NP - 1; i >= 0; --i) {
    const float* row = A + (size_t)i * NP;
    float a0 = 0.0f, a1 = 0.0f;
    for (int j = i + 1 + t; j < NP; j += 256) {
      float uij = row[j];
      a0 += uij * sv[j * 2]; a1 += uij * sv[j * 2 + 1];
    }
    red[2 * t] = a0; red[2 * t + 1] = a1;
    __syncthreads();
    for (int s = 128; s > 0; s >>= 1) {
      if (t < s) { red[2 * t] += red[2 * (t + s)]; red[2 * t + 1] += red[2 * (t + s) + 1]; }
      __syncthreads();
    }
    if (t == 0) {
      float d = row[i];
      sv[i * 2]     = (sv[i * 2]     - red[0]) / d;
      sv[i * 2 + 1] = (sv[i * 2 + 1] - red[1]) / d;
    }
    __syncthreads();
  }
  for (int j = t; j < 2 * NP; j += 256) v[j] = sv[j];
}

// iterative refinement: r = b - (D - W) x  using the *sparse* operator
__global__ void k_res_init(const float* bvec, const float* deg, const int* iverts,
                           const float* x, float* r, int nI, int NP) {
  int i = blockIdx.x * blockDim.x + threadIdx.x;
  if (i >= NP) return;
  if (i < nI) {
    int v = iverts[i];
    float d = deg[v];
    r[i * 2 + 0] = bvec[v * 2 + 0] - d * x[i * 2 + 0];
    r[i * 2 + 1] = bvec[v * 2 + 1] - d * x[i * 2 + 1];
  } else {
    r[i * 2 + 0] = 0.0f; r[i * 2 + 1] = 0.0f;
  }
}

__global__ void k_res_edges(const int* edges, const float* w, const int* imap,
                            const float* x, float* r, int E) {
  int e = blockIdx.x * blockDim.x + threadIdx.x;
  if (e >= E) return;
  int ri = imap[edges[e]], ci = imap[edges[E + e]];
  if (ri >= 0 && ci >= 0) {
    float we = w[e];
    atomicAdd(&r[ri * 2 + 0], we * x[ci * 2 + 0]);
    atomicAdd(&r[ri * 2 + 1], we * x[ci * 2 + 1]);
  }
}

__global__ void k_axpy(float* x, const float* d, int n) {
  int i = blockIdx.x * blockDim.x + threadIdx.x;
  if (i < n) x[i] += d[i];
}

// assemble new_vertices (ws copy + tail of d_out)
__global__ void k_newverts(const int* imap, const float* posf, const float* x,
                           float* nv, float* out_nv, int N) {
  int v = blockIdx.x * blockDim.x + threadIdx.x;
  if (v >= N) return;
  int im = imap[v];
  float a = (im < 0) ? posf[v * 2 + 0] : x[im * 2 + 0];
  float b = (im < 0) ? posf[v * 2 + 1] : x[im * 2 + 1];
  nv[v * 2 + 0] = a; nv[v * 2 + 1] = b;
  out_nv[v * 2 + 0] = a; out_nv[v * 2 + 1] = b;
}

// ---------------------------------------------------------------- streaming warp
__device__ __forceinline__ float cross2h(float px, float py, float qx, float qy) {
  return fabsf(px * qy - py * qx) * 0.5f;
}

__device__ __forceinline__ void warp1(float px, float py, int n0, int n1, int n2,
                                      const float* __restrict__ verts,
                                      const float* __restrict__ nv,
                                      float& ox, float& oy) {
  float2 va = ((const float2*)verts)[n0];
  float2 vb = ((const float2*)verts)[n1];
  float2 vc = ((const float2*)verts)[n2];
  float aA = cross2h(px - vb.x, py - vb.y, px - vc.x, py - vc.y);
  float aB = cross2h(px - va.x, py - va.y, px - vc.x, py - vc.y);
  float aC = cross2h(px - va.x, py - va.y, px - vb.x, py - vb.y);
  float2 na = ((const float2*)nv)[n0];
  float2 nb = ((const float2*)nv)[n1];
  float2 nc = ((const float2*)nv)[n2];
  float inv = 1.0f / (aA + aB + aC);
  ox = (na.x * aA + nb.x * aB + nc.x * aC) * inv;
  oy = (na.y * aA + nb.y * aB + nc.y * aC) * inv;
}

// 2 points per thread: float4 point loads, b128 stores on the 56MB HBM stream
__global__ void __launch_bounds__(256) k_interp(const float* __restrict__ pts,
                                                const int* __restrict__ tri,
                                                const float* __restrict__ verts,
                                                const float* __restrict__ nv,
                                                float* __restrict__ out, int P) {
  int i = blockIdx.x * blockDim.x + threadIdx.x;
  int p = i * 2;
  if (p >= P) return;
  __builtin_prefetch((const char*)(pts + (size_t)p * 2) + 65536, 0, 1);
  __builtin_prefetch((const char*)(tri + (size_t)p * 3) + 98304, 0, 1);

  float4 pp = ((const float4*)pts)[i];
  size_t base = (size_t)p * 3;
  int n0 = tri[base + 0], n1 = tri[base + 1], n2 = tri[base + 2];
  int m0 = tri[base + 3], m1 = tri[base + 4], m2 = tri[base + 5];
  float4 o;
  warp1(pp.x, pp.y, n0, n1, n2, verts, nv, o.x, o.y);
  warp1(pp.z, pp.w, m0, m1, m2, verts, nv, o.z, o.w);
  ((float4*)out)[i] = o;
}

// ---------------------------------------------------------------- host driver
extern "C" void kernel_launch(void* const* d_in, const int* in_sizes, int n_in,
                              void* d_out, int out_size, void* d_ws, size_t ws_size,
                              hipStream_t stream) {
  const float* pts   = (const float*)d_in[0];
  const int*   tri   = (const int*)  d_in[1];
  const float* Wv    = (const float*)d_in[2];
  const float* angv  = (const float*)d_in[3];
  const float* verts = (const float*)d_in[4];
  const int*   edges = (const int*)  d_in[5];
  const int*   bound = (const int*)  d_in[6];
  const int*   iverts= (const int*)  d_in[7];
  const int*   imap  = (const int*)  d_in[8];

  const int P  = in_sizes[0] / 2;
  const int E  = in_sizes[2];
  const int nB = in_sizes[3];
  const int N  = in_sizes[8];
  const int nI = in_sizes[7];
  const int NP = ((nI + 31) / 32) * 32;     // 2432 for R=51

  // workspace carve-up (floats)
  float* ws = (float*)d_ws;
  size_t off = 0;
  auto carve = [&](size_t n) { float* p = ws + off; off += (n + 3) & ~(size_t)3; return p; };
  float* w    = carve(E);
  float* deg  = carve(N);
  float* posf = carve(2 * (size_t)N);
  float* bvec = carve(2 * (size_t)N);
  float* xvec = carve(2 * (size_t)NP);
  float* dr   = carve(2 * (size_t)NP);
  float* nv   = carve(2 * (size_t)N);
  float* A    = carve((size_t)NP * NP);
  if (off * sizeof(float) > ws_size) return;   // not enough scratch

  const int TB = 256;
  auto cdiv = [](int a, int b) { return (a + b - 1) / b; };

  // phase 1: weights / boundary / rhs assembly
  k_zero<<<cdiv(N, TB), TB, 0, stream>>>(deg, N);
  k_zero<<<cdiv(2 * N, TB), TB, 0, stream>>>(posf, 2 * N);
  k_zero<<<cdiv(2 * N, TB), TB, 0, stream>>>(bvec, 2 * N);
  k_weights<<<cdiv(E, TB), TB, 0, stream>>>(Wv, w, E);
  k_boundary<<<1, TB, 0, stream>>>(angv, bound, nB, posf);
  k_deg_b<<<cdiv(E, TB), TB, 0, stream>>>(edges, w, imap, posf, deg, bvec, E);

  // phase 2: dense interior Laplacian (padded, identity on pad rows)
  k_initA<<<2048, TB, 0, stream>>>(A, NP, nI);
  k_scatterA<<<cdiv(E, TB), TB, 0, stream>>>(edges, w, imap, A, NP, E);
  k_diagA<<<cdiv(nI, TB), TB, 0, stream>>>(iverts, deg, A, NP, nI);

  // phase 3: blocked LU (no pivoting; strictly diagonally dominant),
  // bf16-WMMA trailing updates with LDS-staged panels (L2-resident matrix)
  for (int kb = 0; kb < NP / 32; ++kb) {
    int k0 = kb * 32;
    k_panel<<<1, TB, 0, stream>>>(A, NP, k0);
    int rem = NP - k0 - 32;
    if (rem > 0) {
      k_trsmL<<<cdiv(rem, TB), TB, 0, stream>>>(A, NP, k0, rem);
      k_trsmU<<<cdiv(rem, TB), TB, 0, stream>>>(A, NP, k0, rem);
      dim3 g(cdiv(rem, 64), cdiv(rem, 64));
      k_update<<<g, TB, 0, stream>>>(A, NP, k0, rem);
    }
  }

  // phase 4: triangular solves + one sparse-residual refinement step
  size_t shmem = (2 * (size_t)NP + 512) * sizeof(float);
  k_gather_rhs<<<cdiv(NP, TB), TB, 0, stream>>>(bvec, iverts, xvec, nI, NP);
  k_solve<<<1, TB, shmem, stream>>>(A, NP, xvec);
  k_res_init<<<cdiv(NP, TB), TB, 0, stream>>>(bvec, deg, iverts, xvec, dr, nI, NP);
  k_res_edges<<<cdiv(E, TB), TB, 0, stream>>>(edges, w, imap, xvec, dr, E);
  k_solve<<<1, TB, shmem, stream>>>(A, NP, dr);
  k_axpy<<<cdiv(2 * NP, TB), TB, 0, stream>>>(xvec, dr, 2 * NP);

  // phase 5: assemble new_vertices + stream the 2M-point warp
  float* out = (float*)d_out;
  k_newverts<<<cdiv(N, TB), TB, 0, stream>>>(imap, posf, xvec, nv, out + (size_t)P * 2, N);
  k_interp<<<cdiv(P, 2 * TB), TB, 0, stream>>>(pts, tri, verts, nv, out, P);
}
